// CausalSelfAttention_21217138442290
// MI455X (gfx1250) — matrix operand order repaired and checked by reading, not compiled, not run on hardware
//
#include <hip/hip_runtime.h>

// ---------------------------------------------------------------------------
// CDNA5 (gfx1250) causal self-attention, bf16 WMMA everywhere, f32 accumulate.
// B=2, T=2048, C=1024, NH=16, HD=64.
// ---------------------------------------------------------------------------

typedef __bf16 bf16_t;
typedef bf16_t v16bf __attribute__((ext_vector_type(16)));
typedef bf16_t v8bf  __attribute__((ext_vector_type(8)));
typedef float  v8f   __attribute__((ext_vector_type(8)));

#define BB   2
#define TT   2048
#define CC   1024
#define NH   16
#define HD   64
#define BT   (BB * TT)      // 4096 rows
#define C3   (3 * CC)       // 3072

union V16U { v16bf v; v8bf h[2]; };

__device__ __forceinline__ bf16_t f2bf(float x) { return (bf16_t)x; }

// A-fragment (16-bit, 16x32, row-major source [*, lda]):
// lane m = lane&15, half = lane>>4.
// elems 0..7  -> K = k0 + half*8  + (0..7)   (contiguous 16B)
// elems 8..15 -> K = k0 + 16 + half*8 + (0..7) (contiguous 16B)
__device__ __forceinline__ v16bf load_a_frag(const bf16_t* __restrict__ A, int lda,
                                             int row0, int k0, int m, int half) {
    const bf16_t* p = A + (size_t)(row0 + m) * lda + k0 + half * 8;
    V16U u;
    u.h[0] = *(const v8bf*)(p);
    u.h[1] = *(const v8bf*)(p + 16);
    return u.v;
}

// B-fragment (16-bit, 32x16) from N-major (transposed) source Bt[N, K]:
// lane n = lane&15, half = lane>>4; elems 0..7 -> K = k0 + half*8 + (0..7), elems 8..15 -> K = k0 + 16 + half*8 + (0..7)
__device__ __forceinline__ v16bf load_bt_frag(const bf16_t* __restrict__ Bt, int ldb,
                                              int n0, int k0, int n, int half) {
    const bf16_t* p = Bt + (size_t)(n0 + n) * ldb + k0 + half * 8;
    V16U u;
    u.h[0] = *(const v8bf*)(p);
    u.h[1] = *(const v8bf*)(p + 16);
    return u.v;
}

#define WMMA_BF16(a, b, c) \
    __builtin_amdgcn_wmma_f32_16x16x32_bf16(false, (a), false, (b), (short)0, (c), false, false)

// ---------------------------------------------------------------------------
// Precision conversion / weight re-layout kernels
// ---------------------------------------------------------------------------
__global__ void convert_bf16_kernel(const float* __restrict__ src,
                                    bf16_t* __restrict__ dst, int n) {
    int i = blockIdx.x * blockDim.x + threadIdx.x;
    if (i < n) dst[i] = f2bf(src[i]);
}

// src is [R, Cc] row-major f32; dst is [Cc, R] row-major bf16 (i.e. transposed)
__global__ void transpose_bf16_kernel(const float* __restrict__ src,
                                      bf16_t* __restrict__ dst, int R, int Cc) {
    int i = blockIdx.x * blockDim.x + threadIdx.x;
    if (i < R * Cc) {
        int r = i / Cc, c = i % Cc;
        dst[(size_t)c * R + r] = f2bf(src[i]);
    }
}

// ---------------------------------------------------------------------------
// GEMM 1: qkv = x @ w_attn + b_attn, scattered into head-major Q/K and
// transposed V (so the attention kernel's B-fragments are contiguous loads).
// A: [4096,1024] bf16 row-major.  Bt: [3072,1024] bf16 (w_attn transposed).
// Block: 128(M) x 128(N), 256 threads = 8 waves, each wave 32x64.
// Key point: the 128-wide N tile never straddles a 1024 (q/k/v) boundary,
// so the q/k/v selector is UNIFORM per block -> scalar branch, no exec churn.
// ---------------------------------------------------------------------------
__global__ __launch_bounds__(256) void gemm_qkv_kernel(
    const bf16_t* __restrict__ A, const bf16_t* __restrict__ Bt,
    const float* __restrict__ bias,
    bf16_t* __restrict__ qws, bf16_t* __restrict__ kws, bf16_t* __restrict__ vtws) {
    const int K = CC;
    const int lane = threadIdx.x & 31;
    const int wid  = threadIdx.x >> 5;
    const int half = lane >> 4;
    const int nl   = lane & 15;
    const int wm = wid & 3, wn = wid >> 2;
    const int m0 = blockIdx.y * 128 + wm * 32;
    const int n0 = blockIdx.x * 128 + wn * 64;

    v8f acc[2][4];
#pragma unroll
    for (int mi = 0; mi < 2; ++mi)
#pragma unroll
        for (int ni = 0; ni < 4; ++ni) acc[mi][ni] = (v8f){};

    for (int k0 = 0; k0 < K; k0 += 32) {
        v16bf aF[2], bF[4];
#pragma unroll
        for (int mi = 0; mi < 2; ++mi) aF[mi] = load_a_frag(A, K, m0 + mi * 16, k0, nl, half);
#pragma unroll
        for (int ni = 0; ni < 4; ++ni) bF[ni] = load_bt_frag(Bt, K, n0 + ni * 16, k0, nl, half);
#pragma unroll
        for (int mi = 0; mi < 2; ++mi)
#pragma unroll
            for (int ni = 0; ni < 4; ++ni)
                acc[mi][ni] = WMMA_BF16(aF[mi], bF[ni], acc[mi][ni]);
    }

    // ---- epilogue -----------------------------------------------------------
    // UNIFORM selector: 0 = q, 1 = k, 2 = v (scalar, from blockIdx only)
    const int which = (int)(blockIdx.x * 128) / CC;

    float bv[4];
#pragma unroll
    for (int ni = 0; ni < 4; ++ni) bv[ni] = bias[n0 + ni * 16 + nl];

    // column within the selected 1024-wide slab (per-lane, but branch-free)
    const int c0 = (n0 & (CC - 1)) + nl;   // + ni*16 below
    // row decomposition helpers: row = m0 + mi*16 + r + 8*half -> (b, t)
    const int rbase = m0 + half * 8;       // + mi*16 + r below

    if (which < 2) {
        bf16_t* __restrict__ dst = (which == 0) ? qws : kws;
        // layout: dst[((b*NH + h)*TT + t)*HD + d],  h = c>>6, d = c&63
#pragma unroll
        for (int ni = 0; ni < 4; ++ni) {
            const int c = c0 + ni * 16;
            const int h = c >> 6, d = c & (HD - 1);
#pragma unroll
            for (int mi = 0; mi < 2; ++mi) {
#pragma unroll
                for (int r = 0; r < 8; ++r) {
                    const int row = rbase + mi * 16 + r;
                    const int b = row >> 11, t = row & (TT - 1);
                    const float v = acc[mi][ni][r] + bv[ni];
                    dst[((size_t)(b * NH + h) * TT + t) * HD + d] = f2bf(v);
                }
            }
        }
    } else {
        // layout: vtws[((b*NH + h)*HD + d)*TT + t]
#pragma unroll
        for (int ni = 0; ni < 4; ++ni) {
            const int c = c0 + ni * 16;
            const int h = c >> 6, d = c & (HD - 1);
#pragma unroll
            for (int mi = 0; mi < 2; ++mi) {
#pragma unroll
                for (int r = 0; r < 8; ++r) {
                    const int row = rbase + mi * 16 + r;
                    const int b = row >> 11, t = row & (TT - 1);
                    const float v = acc[mi][ni][r] + bv[ni];
                    vtws[((size_t)(b * NH + h) * HD + d) * TT + t] = f2bf(v);
                }
            }
        }
    }
}

// ---------------------------------------------------------------------------
// GEMM 2: out = y @ w_proj + b_proj, f32 output (final result).
// ---------------------------------------------------------------------------
__global__ __launch_bounds__(256) void gemm_out_kernel(
    const bf16_t* __restrict__ A, const bf16_t* __restrict__ Bt,
    const float* __restrict__ bias, float* __restrict__ out) {
    const int K = CC;
    const int lane = threadIdx.x & 31;
    const int wid  = threadIdx.x >> 5;
    const int half = lane >> 4;
    const int nl   = lane & 15;
    const int wm = wid & 3, wn = wid >> 2;
    const int m0 = blockIdx.y * 128 + wm * 32;
    const int n0 = blockIdx.x * 128 + wn * 64;

    v8f acc[2][4];
#pragma unroll
    for (int mi = 0; mi < 2; ++mi)
#pragma unroll
        for (int ni = 0; ni < 4; ++ni) acc[mi][ni] = (v8f){};

    for (int k0 = 0; k0 < K; k0 += 32) {
        v16bf aF[2], bF[4];
#pragma unroll
        for (int mi = 0; mi < 2; ++mi) aF[mi] = load_a_frag(A, K, m0 + mi * 16, k0, nl, half);
#pragma unroll
        for (int ni = 0; ni < 4; ++ni) bF[ni] = load_bt_frag(Bt, K, n0 + ni * 16, k0, nl, half);
#pragma unroll
        for (int mi = 0; mi < 2; ++mi)
#pragma unroll
            for (int ni = 0; ni < 4; ++ni)
                acc[mi][ni] = WMMA_BF16(aF[mi], bF[ni], acc[mi][ni]);
    }

    float bv[4];
#pragma unroll
    for (int ni = 0; ni < 4; ++ni) bv[ni] = bias[n0 + ni * 16 + nl];

#pragma unroll
    for (int mi = 0; mi < 2; ++mi)
#pragma unroll
        for (int r = 0; r < 8; ++r) {
            const int row = m0 + mi * 16 + r + half * 8;
            float* __restrict__ orow = out + (size_t)row * CC + n0 + nl;
#pragma unroll
            for (int ni = 0; ni < 4; ++ni)
                orow[ni * 16] = acc[mi][ni][r] + bv[ni];
        }
}

// ---------------------------------------------------------------------------
// Flash attention: one wave owns 16 query rows of one (b, h).
// Per iteration: 32 keys -> two 16x16 score tiles (QK^T via 2x2 WMMAs),
// scale + causal mask, online softmax (shfl_xor reductions within 16-lane
// halves, since row m = r + 8*half lives across lanes (half*16 .. half*16+15)),
// P retiled through LDS into the bf16 A-fragment layout, then 4 WMMAs of P@V.
// ---------------------------------------------------------------------------
__global__ __launch_bounds__(256) void attn_kernel(
    const bf16_t* __restrict__ qws, const bf16_t* __restrict__ kws,
    const bf16_t* __restrict__ vtws, bf16_t* __restrict__ yws) {
    __shared__ bf16_t lds_p[8][16 * 32];   // per-wave 16x32 P staging (8 KB)

    const int lane = threadIdx.x & 31;
    const int wid  = threadIdx.x >> 5;
    const int half = lane >> 4;
    const int nl   = lane & 15;

    const int gw = blockIdx.x * 8 + wid;   // 0 .. B*NH*(T/16)-1 = 4095
    const int qt = gw & (TT / 16 - 1);     // query tile within (b,h)
    const int bh = gw >> 7;                // b*NH + h
    const int q0 = qt * 16;

    const bf16_t* qb = qws  + (size_t)bh * TT * HD;
    const bf16_t* kb = kws  + (size_t)bh * TT * HD;
    const bf16_t* vb = vtws + (size_t)bh * HD * TT;

    v16bf qA[2];
#pragma unroll
    for (int kk = 0; kk < 2; ++kk) qA[kk] = load_a_frag(qb, HD, q0, kk * 32, nl, half);

    v8f o0 = (v8f){}, o1 = (v8f){}, o2 = (v8f){}, o3 = (v8f){};
    float mstat[8], lstat[8];
#pragma unroll
    for (int r = 0; r < 8; ++r) { mstat[r] = -3.0e38f; lstat[r] = 0.0f; }

    const float scale = 0.125f;  // 1/sqrt(HD)

    for (int j0 = 0; j0 < q0 + 16; j0 += 32) {
        // ---- scores: two 16x16 tiles over K = HD = 64 (2 wmma steps each)
        v8f s[2];
#pragma unroll
        for (int jj = 0; jj < 2; ++jj) {
            v8f t = (v8f){};
#pragma unroll
            for (int kk = 0; kk < 2; ++kk) {
                v16bf kF = load_bt_frag(kb, HD, j0 + jj * 16, kk * 32, nl, half);
                t = WMMA_BF16(qA[kk], kF, t);
            }
            s[jj] = t;
        }
        // ---- scale + causal mask (D elem r: row = q0 + r + 8*half, col = tile + nl)
#pragma unroll
        for (int jj = 0; jj < 2; ++jj)
#pragma unroll
            for (int r = 0; r < 8; ++r) {
                int mrow = q0 + r + half * 8;
                int ncol = j0 + jj * 16 + nl;
                float v = s[jj][r] * scale;
                s[jj][r] = (ncol <= mrow) ? v : -3.0e38f;
            }
        // ---- online softmax per row
#pragma unroll
        for (int r = 0; r < 8; ++r) {
            float rm = fmaxf(s[0][r], s[1][r]);
#pragma unroll
            for (int off = 1; off < 16; off <<= 1) rm = fmaxf(rm, __shfl_xor(rm, off, 32));
            float mnew = fmaxf(mstat[r], rm);
            float p0 = __expf(s[0][r] - mnew);
            float p1 = __expf(s[1][r] - mnew);
            float rs = p0 + p1;
#pragma unroll
            for (int off = 1; off < 16; off <<= 1) rs += __shfl_xor(rs, off, 32);
            float corr = __expf(mstat[r] - mnew);
            lstat[r] = lstat[r] * corr + rs;
            mstat[r] = mnew;
            s[0][r] = p0; s[1][r] = p1;
            o0[r] *= corr; o1[r] *= corr; o2[r] *= corr; o3[r] *= corr;
        }
        // ---- P (C/D layout) -> LDS row-major 16x32 -> bf16 A-fragment
        bf16_t* lp = lds_p[wid];
#pragma unroll
        for (int jj = 0; jj < 2; ++jj)
#pragma unroll
            for (int r = 0; r < 8; ++r)
                lp[(r + half * 8) * 32 + jj * 16 + nl] = f2bf(s[jj][r]);
        // same-wave RAW through LDS: compiler inserts the ds-cnt wait
        v16bf pA;
        {
            const bf16_t* p = lp + nl * 32 + half * 8;  // A-frag layout, lda=32
            V16U u;
            u.h[0] = *(const v8bf*)(p);
            u.h[1] = *(const v8bf*)(p + 16);
            pA = u.v;
        }
        // ---- O += P @ V  (V transposed [HD, T] -> contiguous B-fragments)
        {
            v16bf vF;
            vF = load_bt_frag(vb, TT, 0,  j0, nl, half); o0 = WMMA_BF16(pA, vF, o0);
            vF = load_bt_frag(vb, TT, 16, j0, nl, half); o1 = WMMA_BF16(pA, vF, o1);
            vF = load_bt_frag(vb, TT, 32, j0, nl, half); o2 = WMMA_BF16(pA, vF, o2);
            vF = load_bt_frag(vb, TT, 48, j0, nl, half); o3 = WMMA_BF16(pA, vF, o3);
        }
    }

    // ---- normalize and write y in [B, T, C] layout (bf16 for the final GEMM)
    const int b = bh >> 4, h = bh & (NH - 1);
#pragma unroll
    for (int r = 0; r < 8; ++r) {
        int trow = q0 + r + half * 8;
        float inv = 1.0f / lstat[r];
        bf16_t* __restrict__ yrow = yws + ((size_t)(b * TT + trow)) * CC + h * HD + nl;
        yrow[0]  = f2bf(o0[r] * inv);
        yrow[16] = f2bf(o1[r] * inv);
        yrow[32] = f2bf(o2[r] * inv);
        yrow[48] = f2bf(o3[r] * inv);
    }
}

// ---------------------------------------------------------------------------
// Launch
// ---------------------------------------------------------------------------
extern "C" void kernel_launch(void* const* d_in, const int* in_sizes, int n_in,
                              void* d_out, int out_size, void* d_ws, size_t ws_size,
                              hipStream_t stream) {
    const float* x      = (const float*)d_in[0];   // [B,T,C]
    const float* w_attn = (const float*)d_in[1];   // [C,3C]
    const float* b_attn = (const float*)d_in[2];   // [3C]
    const float* w_proj = (const float*)d_in[3];   // [C,C]
    const float* b_proj = (const float*)d_in[4];   // [C]
    float* out = (float*)d_out;

    char* ws = (char*)d_ws;
    size_t off = 0;
    bf16_t* x_bf    = (bf16_t*)(ws + off); off += (size_t)BT * CC * 2;       // 8 MB
    bf16_t* wattnT  = (bf16_t*)(ws + off); off += (size_t)C3 * CC * 2;       // 6 MB
    bf16_t* wprojT  = (bf16_t*)(ws + off); off += (size_t)CC * CC * 2;       // 2 MB
    bf16_t* q_ws    = (bf16_t*)(ws + off); off += (size_t)BB * NH * TT * HD * 2; // 8 MB
    bf16_t* k_ws    = (bf16_t*)(ws + off); off += (size_t)BB * NH * TT * HD * 2; // 8 MB
    bf16_t* vT_ws   = (bf16_t*)(ws + off); off += (size_t)BB * NH * HD * TT * 2; // 8 MB
    bf16_t* y_ws    = (bf16_t*)(ws + off); off += (size_t)BT * CC * 2;       // 8 MB
    (void)ws_size; (void)in_sizes; (void)n_in; (void)out_size;

    // 1) precision conversion / weight re-layout
    {
        int n = BT * CC;
        convert_bf16_kernel<<<(n + 255) / 256, 256, 0, stream>>>(x, x_bf, n);
    }
    {
        int n = CC * C3;
        transpose_bf16_kernel<<<(n + 255) / 256, 256, 0, stream>>>(w_attn, wattnT, CC, C3);
    }
    {
        int n = CC * CC;
        transpose_bf16_kernel<<<(n + 255) / 256, 256, 0, stream>>>(w_proj, wprojT, CC, CC);
    }

    // 2) qkv projection, scattered to head-major Q/K and transposed V
    gemm_qkv_kernel<<<dim3(C3 / 128, BT / 128), 256, 0, stream>>>(
        x_bf, wattnT, b_attn, q_ws, k_ws, vT_ws);

    // 3) flash attention: B*NH*(T/16) = 4096 waves, 8 waves per block
    attn_kernel<<<dim3((BB * NH * (TT / 16)) / 8), 256, 0, stream>>>(
        q_ws, k_ws, vT_ws, y_ws);

    // 4) output projection -> f32 result
    gemm_out_kernel<<<dim3(CC / 128, BT / 128), 256, 0, stream>>>(
        y_ws, wprojT, b_proj, out);
}
